// GraphHead_21835613733001
// MI455X (gfx1250) — matrix operand order repaired and checked by reading, not compiled, not run on hardware
//
#include <hip/hip_runtime.h>
#include <hip/hip_bf16.h>

typedef __attribute__((ext_vector_type(16))) __bf16 v16bf;
typedef __attribute__((ext_vector_type(8)))  __bf16 v8bf;
typedef __attribute__((ext_vector_type(8)))  float  v8f;

#define CL 64
#define NN 8

// ---------------- LDS layout (bytes) ----------------
#define OFF_WT_ENC   0              // 64x64 bf16 (transposed: [o][k]; cols 0,1 patched to identity)
#define OFF_WT_S0    (OFF_WT_ENC  + 8192)
#define OFF_WT_S1    (OFF_WT_S0   + 8192)
#define OFF_WPROJ    (OFF_WT_S1   + 8192)   // combined 512x64 bf16: [WaT_rel|WbT_rel|WaT_att|WbT_att]
#define OFF_WT_REL1  (OFF_WPROJ   + 65536)  // 64x128 bf16
#define OFF_WT_REL2  (OFF_WT_REL1 + 16384)  // 64x64
#define OFF_WT_ATT1  (OFF_WT_REL2 + 8192)   // 64x128
#define OFF_ACT_S    (OFF_WT_ATT1 + 16384)  // 16x64 bf16, rows 8..15 zero
#define OFF_ACT_SP   (OFF_ACT_S   + 2048)
#define OFF_ACT_H    (OFF_ACT_SP  + 2048)
#define OFF_PROJ     (OFF_ACT_H   + 2048)   // 8 nodes x 512 bf16: [pArel|pBrel|pAatt|pBatt]
#define OFF_PW       (OFF_PROJ    + 8192)   // 4 waves x 16x64 bf16 scratch
#define OFF_SMALLF   (OFF_PW      + 8192)   // 1280 f32 (biases, Wd, w_att2, bproj)
#define OFF_POS      (OFF_SMALLF  + 5120)   // 16 f32
#define OFF_SDSUM    (OFF_POS     + 64)     // 64 f32
#define OFF_OUTACC   (OFF_SDSUM   + 256)    // 4x64 f32
#define OFF_ATTV     (OFF_OUTACC  + 1024)   // 4x16 f32
#define SMEM_BYTES   (OFF_ATTV    + 256)    // ~156.6 KB -> 2 blocks / WGP

// f32 offsets inside smallf (elements)
#define F_BENC   0
#define F_BS0    64
#define F_BS1    128
#define F_WDREL  192
#define F_BREL1  320
#define F_BREL2  384
#define F_WDATT  448
#define F_BATT1  576
#define F_WATT2  640
#define F_BATT2  704
#define F_BPROJ  768   // 512 entries: [b_rel0 | 0 | b_att0 | 0]

__device__ __forceinline__ float eluf(float x) {
  return x > 0.f ? x : (__expf(x) - 1.f);
}

__device__ __forceinline__ v8f wmma_bf16(v16bf a, v16bf b, v8f c) {
  return __builtin_amdgcn_wmma_f32_16x16x32_bf16(false, a, false, b, (short)0, c, false, false);
}

// A-tile (16x32 bf16) from row-major bf16 activation buffer (stride elems).
// lane L: row m=L&15; hi=L>>4. elem e -> K = ((e>>3)<<4) + hi*8 + (e&7).
__device__ __forceinline__ v16bf load_A(const __bf16* act, int stride, int k0, int lane) {
  const int m = lane & 15, hi = lane >> 4;
  const __bf16* p0 = act + m * stride + k0 + hi * 8;
  const __bf16* p1 = act + m * stride + k0 + 16 + hi * 8;
  v8bf g0 = *(const v8bf*)p0;
  v8bf g1 = *(const v8bf*)p1;
  return __builtin_shufflevector(g0, g1, 0,1,2,3,4,5,6,7,8,9,10,11,12,13,14,15);
}

// B-tile (32x16 bf16) from transposed weights WT[o][Kdim] bf16 in LDS.
// lane L: col n = n0 + (L&15); elem e -> K = k0 + ((e>>3)<<4) + (L>>4)*8 + (e&7).
__device__ __forceinline__ v16bf load_B(const __bf16* wt, int Kdim, int n0, int k0, int lane) {
  const int n = lane & 15, hi = lane >> 4;
  const __bf16* p = wt + (n0 + n) * Kdim + k0 + hi * 8;
  v8bf g0 = *(const v8bf*)p;
  v8bf g1 = *(const v8bf*)(p + 16);
  return __builtin_shufflevector(g0, g1, 0,1,2,3,4,5,6,7,8,9,10,11,12,13,14,15);
}

// Build pair-MLP input A-tile: X = elu(pA[i] + pB[j] + dist*Wd)  (bias pre-folded into pA)
__device__ __forceinline__ v16bf build_X(const __bf16* proj, const float* wd,
                                         int i, int j, float dist,
                                         int offA, int offB, int k0, int lane) {
  const int hi = lane >> 4;
  const int kb0 = k0 + hi * 8;
  const int kb1 = k0 + 16 + hi * 8;
  v8bf pa0 = *(const v8bf*)(proj + i * 512 + offA + kb0);
  v8bf pb0 = *(const v8bf*)(proj + j * 512 + offB + kb0);
  v8bf pa1 = *(const v8bf*)(proj + i * 512 + offA + kb1);
  v8bf pb1 = *(const v8bf*)(proj + j * 512 + offB + kb1);
  v16bf r;
#pragma unroll
  for (int e = 0; e < 8; ++e) {
    float x0 = (float)pa0[e] + (float)pb0[e] + dist * wd[kb0 + e];
    float x1 = (float)pa1[e] + (float)pb1[e] + dist * wd[kb1 + e];
    r[e]     = (__bf16)eluf(x0);
    r[8 + e] = (__bf16)eluf(x1);
  }
  return r;
}

// Stage fp32 weight [K][O] (row-major, O fastest) -> transposed bf16 dst[o*K + k]
__device__ __forceinline__ void stageT(__bf16* dst, const float* W, int K, int O,
                                       int tid, int nth) {
  for (int idx = tid; idx < K * O; idx += nth) {
    int o = idx / K, k = idx - o * K;
    dst[idx] = (__bf16)W[k * O + o];
  }
}

__device__ __forceinline__ void copyf(float* dst, const float* src, int n, int tid, int nth) {
  for (int i = tid; i < n; i += nth) dst[i] = src[i];
}

__global__ __launch_bounds__(128)
void graphhead_fused(const float* __restrict__ s,
                     const float* __restrict__ W_enc,  const float* __restrict__ b_enc,
                     const float* __restrict__ W_s0,   const float* __restrict__ b_s0,
                     const float* __restrict__ W_s1,   const float* __restrict__ b_s1,
                     const float* __restrict__ W_rel0, const float* __restrict__ b_rel0,
                     const float* __restrict__ W_rel1, const float* __restrict__ b_rel1,
                     const float* __restrict__ W_rel2, const float* __restrict__ b_rel2,
                     const float* __restrict__ W_att0, const float* __restrict__ b_att0,
                     const float* __restrict__ W_att1, const float* __restrict__ b_att1,
                     const float* __restrict__ W_att2, const float* __restrict__ b_att2,
                     float* __restrict__ out, int Btot) {
  extern __shared__ __align__(16) char smem[];
  __bf16* WT_enc  = (__bf16*)(smem + OFF_WT_ENC);
  __bf16* WT_s0   = (__bf16*)(smem + OFF_WT_S0);
  __bf16* WT_s1   = (__bf16*)(smem + OFF_WT_S1);
  __bf16* Wproj   = (__bf16*)(smem + OFF_WPROJ);   // 512x64, rows: gcol of [pArel|pBrel|pAatt|pBatt]
  __bf16* WT_rel1 = (__bf16*)(smem + OFF_WT_REL1);
  __bf16* WT_rel2 = (__bf16*)(smem + OFF_WT_REL2);
  __bf16* WT_att1 = (__bf16*)(smem + OFF_WT_ATT1);
  __bf16* act_s   = (__bf16*)(smem + OFF_ACT_S);
  __bf16* act_sp  = (__bf16*)(smem + OFF_ACT_SP);
  __bf16* act_h   = (__bf16*)(smem + OFF_ACT_H);
  __bf16* proj    = (__bf16*)(smem + OFF_PROJ);
  __bf16* pwbase  = (__bf16*)(smem + OFF_PW);
  float*  smallf  = (float*)(smem + OFF_SMALLF);
  float*  pos     = (float*)(smem + OFF_POS);
  float*  sd_sum  = (float*)(smem + OFF_SDSUM);
  float*  out_acc = (float*)(smem + OFF_OUTACC);
  float*  attv    = (float*)(smem + OFF_ATTV);

  const int tid  = threadIdx.x;
  const int w    = tid >> 5;
  const int lane = tid & 31;
  const int m    = lane & 15;
  const int hi   = lane >> 4;

  // ---- one-time per-block staging: weights -> bf16 transposed in LDS ----
  // Encoder: columns 0,1 replaced by identity rows so the GEMM itself performs
  // the concat(s[...,:2], enc[...,2:]) — no per-element select needed later.
  for (int idx = tid; idx < 64 * 64; idx += 128) {
    int o = idx >> 6, k = idx & 63;
    float v = W_enc[k * 64 + o];
    if (o < 2) v = (k == o) ? 1.f : 0.f;
    WT_enc[idx] = (__bf16)v;
  }
  stageT(WT_s0,   W_s0,             64, 64,  tid, 128);
  stageT(WT_s1,   W_s1,             64, 64,  tid, 128);
  stageT(Wproj + 0 * 8192, W_rel0,          64, 128, tid, 128);  // WaT_rel
  stageT(Wproj + 1 * 8192, W_rel0 + 64*128, 64, 128, tid, 128);  // WbT_rel
  stageT(Wproj + 2 * 8192, W_att0,          64, 128, tid, 128);  // WaT_att
  stageT(Wproj + 3 * 8192, W_att0 + 64*128, 64, 128, tid, 128);  // WbT_att
  stageT(WT_rel1, W_rel1,           128, 64, tid, 128);
  stageT(WT_rel2, W_rel2,           64, 64,  tid, 128);
  stageT(WT_att1, W_att1,           128, 64, tid, 128);
  for (int i = tid; i < 64; i += 128)   // b_enc with passthrough cols zeroed
    smallf[F_BENC + i] = (i < 2) ? 0.f : b_enc[i];
  copyf(smallf + F_BS0,   b_s0,   64,  tid, 128);
  copyf(smallf + F_BS1,   b_s1,   64,  tid, 128);
  copyf(smallf + F_WDREL, W_rel0 + 128*128, 128, tid, 128);
  copyf(smallf + F_BREL1, b_rel1, 64,  tid, 128);
  copyf(smallf + F_BREL2, b_rel2, 64,  tid, 128);
  copyf(smallf + F_WDATT, W_att0 + 128*128, 128, tid, 128);
  copyf(smallf + F_BATT1, b_att1, 64,  tid, 128);
  copyf(smallf + F_WATT2, W_att2, 64,  tid, 128);
  if (tid == 0) smallf[F_BATT2] = b_att2[0];
  // combined projection bias: [b_rel0 | 0 | b_att0 | 0]
  for (int i = tid; i < 512; i += 128) {
    float v = 0.f;
    if (i < 128) v = b_rel0[i];
    else if (i >= 256 && i < 384) v = b_att0[i - 256];
    smallf[F_BPROJ + i] = v;
  }
  // zero padding rows 8..15 of the 16-row activation staging buffers (stay zero)
  for (int idx = tid; idx < 8 * 64; idx += 128) {
    act_s [512 + idx] = (__bf16)0.f;
    act_sp[512 + idx] = (__bf16)0.f;
    act_h [512 + idx] = (__bf16)0.f;
  }
  __syncthreads();

  for (int b = blockIdx.x; b < Btot; b += gridDim.x) {
    // prefetch next iteration's s tile (gfx1250 global_prefetch)
    const int bn = b + gridDim.x;
    if (bn < Btot) __builtin_prefetch(s + (size_t)bn * NN * CL + tid * 4, 0, 0);

    // ---- stage 1: load s[b] (8x64) into LDS bf16 + fp32 positions ----
    {
      const float4 v = ((const float4*)(s + (size_t)b * NN * CL))[tid];
      const int base = tid * 4;
      const int node = base >> 6, c0 = base & 63;
#pragma unroll
      for (int e = 0; e < 4; ++e)
        act_s[node * 64 + c0 + e] = (__bf16)(&v.x)[e];
      if (tid < 16)
        pos[tid] = s[(size_t)b * NN * CL + (tid >> 1) * 64 + (tid & 1)];
    }
    __syncthreads();

    // ---- stage 2: encoder (wave w owns output cols [16w,16w+16)) ----
    {
      v16bf A0 = load_A(act_s, 64, 0, lane);
      v16bf A1 = load_A(act_s, 64, 32, lane);
      v8f c = {0.f,0.f,0.f,0.f,0.f,0.f,0.f,0.f};
      c = wmma_bf16(A0, load_B(WT_enc, 64, 16 * w, 0, lane), c);
      c = wmma_bf16(A1, load_B(WT_enc, 64, 16 * w, 32, lane), c);
      const int col = 16 * w + m;
      if (hi == 0) {
        const float benc = smallf[F_BENC + col];  // 0 for passthrough cols
#pragma unroll
        for (int r = 0; r < 8; ++r)
          act_sp[r * 64 + col] = (__bf16)(c[r] + benc);  // uniform, branch-free
      }
    }
    __syncthreads();

    // ---- stage 3: self0 + the four pair projections (from s') ----
    float hreg[8];   // h = elu(s'@W_s0 + b_s0) for rows 0..7 at this lane's col
    {
      v16bf A0 = load_A(act_sp, 64, 0, lane);
      v16bf A1 = load_A(act_sp, 64, 32, lane);
      const int col = 16 * w + m;
      {
        v8f c = {0.f,0.f,0.f,0.f,0.f,0.f,0.f,0.f};
        c = wmma_bf16(A0, load_B(WT_s0, 64, 16 * w, 0, lane), c);
        c = wmma_bf16(A1, load_B(WT_s0, 64, 16 * w, 32, lane), c);
        const float bs0 = smallf[F_BS0 + col];
#pragma unroll
        for (int r = 0; r < 8; ++r)
          hreg[r] = eluf(c[r] + bs0);
        if (hi == 0) {
#pragma unroll
          for (int r = 0; r < 8; ++r)
            act_h[r * 64 + col] = (__bf16)hreg[r];
        }
      }
      // projections: combined 512-wide output [pArel|pBrel|pAatt|pBatt], 32 N-tiles,
      // Wproj is one contiguous 512x64 transposed matrix -> fully uniform loop
#pragma unroll
      for (int tt = 0; tt < 8; ++tt) {
        const int t = w + 4 * tt;
        v8f cp = {0.f,0.f,0.f,0.f,0.f,0.f,0.f,0.f};
        cp = wmma_bf16(A0, load_B(Wproj, 64, 16 * t, 0, lane), cp);
        cp = wmma_bf16(A1, load_B(Wproj, 64, 16 * t, 32, lane), cp);
        const int gcol = 16 * t + m;
        const float bias = smallf[F_BPROJ + gcol];
        if (hi == 0) {
#pragma unroll
          for (int r = 0; r < 8; ++r)
            proj[r * 512 + gcol] = (__bf16)(cp[r] + bias);
        }
      }
    }
    __syncthreads();

    // ---- stage 4: self1 + residual (h kept in regs), column-summed over nodes ----
    {
      v16bf A0 = load_A(act_h, 64, 0, lane);
      v16bf A1 = load_A(act_h, 64, 32, lane);
      v8f c = {0.f,0.f,0.f,0.f,0.f,0.f,0.f,0.f};
      c = wmma_bf16(A0, load_B(WT_s1, 64, 16 * w, 0, lane), c);
      c = wmma_bf16(A1, load_B(WT_s1, 64, 16 * w, 32, lane), c);
      const int col = 16 * w + m;
      if (hi == 0) {
        const float cb = smallf[F_BS1 + col];
        float ssum = 0.f;
#pragma unroll
        for (int r = 0; r < 8; ++r)
          ssum += c[r] + cb + hreg[r];
        sd_sum[col] = ssum;
      }
    }

    // ---- stage 5: pair phase — wave w owns pairs p = 16w..16w+15 ----
    out_acc[w * 64 + lane] = 0.f;
    out_acc[w * 64 + 32 + lane] = 0.f;
    __bf16* pw = pwbase + w * 1024;
    const int p_ln = 16 * w + m;
    const int i_ln = p_ln >> 3, j_ln = p_ln & 7;
    const float dx = pos[2 * i_ln] - pos[2 * j_ln];
    const float dy = pos[2 * i_ln + 1] - pos[2 * j_ln + 1];
    const float dist = dx * dx + dy * dy;

    // attention path
    {
      v16bf Xa[4];
#pragma unroll
      for (int kk = 0; kk < 4; ++kk)
        Xa[kk] = build_X(proj, smallf + F_WDATT, i_ln, j_ln, dist, 256, 384, 32 * kk, lane);
#pragma unroll
      for (int t = 0; t < 4; ++t) {
        v8f c = {0.f,0.f,0.f,0.f,0.f,0.f,0.f,0.f};
#pragma unroll
        for (int kk = 0; kk < 4; ++kk)
          c = wmma_bf16(Xa[kk], load_B(WT_att1, 128, 16 * t, 32 * kk, lane), c);
        const int col = 16 * t + m;
        const float bias = smallf[F_BATT1 + col];
#pragma unroll
        for (int r = 0; r < 8; ++r)
          pw[(r + 8 * hi) * 64 + col] = (__bf16)eluf(c[r] + bias);
      }
      // per-row 64-wide dot with w_att2, split across the two half-waves
      {
        float acc = 0.f;
        const int kbase = hi * 32;
#pragma unroll
        for (int k8 = 0; k8 < 32; k8 += 8) {
          v8bf hh = *(const v8bf*)(pw + m * 64 + kbase + k8);
#pragma unroll
          for (int e = 0; e < 8; ++e)
            acc += (float)hh[e] * smallf[F_WATT2 + kbase + k8 + e];
        }
        acc += __shfl_down(acc, 16, 32);
        if (hi == 0)
          attv[w * 16 + m] = 1.f / (1.f + __expf(-(acc + smallf[F_BATT2])));
      }
    }

    // relation path
    {
      v16bf Xr[4];
#pragma unroll
      for (int kk = 0; kk < 4; ++kk)
        Xr[kk] = build_X(proj, smallf + F_WDREL, i_ln, j_ln, dist, 0, 128, 32 * kk, lane);
      float h2f[4][8];  // rel_h2 values this lane owns (rows r+8*hi, col 16t+m)
#pragma unroll
      for (int t = 0; t < 4; ++t) {
        v8f c = {0.f,0.f,0.f,0.f,0.f,0.f,0.f,0.f};
#pragma unroll
        for (int kk = 0; kk < 4; ++kk)
          c = wmma_bf16(Xr[kk], load_B(WT_rel1, 128, 16 * t, 32 * kk, lane), c);
        const int col = 16 * t + m;
        const float bias = smallf[F_BREL1 + col];
#pragma unroll
        for (int r = 0; r < 8; ++r) {
          h2f[t][r] = eluf(c[r] + bias);
          pw[(r + 8 * hi) * 64 + col] = (__bf16)h2f[t][r];  // rel_h2 for A-tiles
        }
      }
      // per-row weights: att * (i != j)
      float wrow[8];
#pragma unroll
      for (int r = 0; r < 8; ++r) {
        const int mr = r + 8 * hi, pp = 16 * w + mr;
        const int ii = pp >> 3, jj = pp & 7;
        wrow[r] = (ii == jj) ? 0.f : attv[w * 16 + mr];
      }
      v16bf A20 = load_A(pw, 64, 0, lane);
      v16bf A21 = load_A(pw, 64, 32, lane);
#pragma unroll
      for (int t = 0; t < 4; ++t) {
        v8f c = {0.f,0.f,0.f,0.f,0.f,0.f,0.f,0.f};
        c = wmma_bf16(A20, load_B(WT_rel2, 64, 16 * t, 0, lane), c);
        c = wmma_bf16(A21, load_B(WT_rel2, 64, 16 * t, 32, lane), c);
        const int col = 16 * t + m;
        const float b2 = smallf[F_BREL2 + col];
        float sum = 0.f;
#pragma unroll
        for (int r = 0; r < 8; ++r)
          sum += (c[r] + b2 + h2f[t][r]) * wrow[r];  // rel_f * mask * att
        sum += __shfl_down(sum, 16, 32);             // merge the two row-halves
        if (hi == 0) out_acc[w * 64 + col] += sum;   // per-wave, single writer
      }
    }
    __syncthreads();

    if (tid < 64) {
      float v = sd_sum[tid] + out_acc[tid] + out_acc[64 + tid] +
                out_acc[128 + tid] + out_acc[192 + tid];
      out[(size_t)b * 64 + tid] = v;
    }
    __syncthreads();
  }
}

extern "C" void kernel_launch(void* const* d_in, const int* in_sizes, int n_in,
                              void* d_out, int out_size, void* d_ws, size_t ws_size,
                              hipStream_t stream) {
  (void)n_in; (void)out_size; (void)d_ws; (void)ws_size;
  const float* s      = (const float*)d_in[0];
  const float* W_enc  = (const float*)d_in[1];
  const float* b_enc  = (const float*)d_in[2];
  const float* W_s0   = (const float*)d_in[3];
  const float* b_s0   = (const float*)d_in[4];
  const float* W_s1   = (const float*)d_in[5];
  const float* b_s1   = (const float*)d_in[6];
  const float* W_rel0 = (const float*)d_in[7];
  const float* b_rel0 = (const float*)d_in[8];
  const float* W_rel1 = (const float*)d_in[9];
  const float* b_rel1 = (const float*)d_in[10];
  const float* W_rel2 = (const float*)d_in[11];
  const float* b_rel2 = (const float*)d_in[12];
  const float* W_att0 = (const float*)d_in[13];
  const float* b_att0 = (const float*)d_in[14];
  const float* W_att1 = (const float*)d_in[15];
  const float* b_att1 = (const float*)d_in[16];
  const float* W_att2 = (const float*)d_in[17];
  const float* b_att2 = (const float*)d_in[18];
  float* out = (float*)d_out;

  const int Btot = in_sizes[0] / (NN * CL);
  hipFuncSetAttribute((const void*)graphhead_fused,
                      hipFuncAttributeMaxDynamicSharedMemorySize, SMEM_BYTES);
  int blocks = Btot < 1024 ? Btot : 1024;
  graphhead_fused<<<dim3(blocks), dim3(128), SMEM_BYTES, stream>>>(
      s, W_enc, b_enc, W_s0, b_s0, W_s1, b_s1,
      W_rel0, b_rel0, W_rel1, b_rel1, W_rel2, b_rel2,
      W_att0, b_att0, W_att1, b_att1, W_att2, b_att2,
      out, Btot);
}
